// GraphAttentionLayer_24086176596112
// MI455X (gfx1250) — matrix operand order repaired and checked by reading, not compiled, not run on hardware
//
#include <hip/hip_runtime.h>

typedef __attribute__((ext_vector_type(16))) _Float16 v16h;
typedef __attribute__((ext_vector_type(8)))  _Float16 v8h;
typedef __attribute__((ext_vector_type(8)))  float    v8f;
typedef __attribute__((ext_vector_type(4)))  int      v4i;
typedef __attribute__((ext_vector_type(4)))  float    v4f;

#define NN      8192
#define IN_F    128
#define OUT_F   64
#define LALPHA  0.2f
#define LOG2E   1.44269504088896f
#define NSPLIT  8
#define CPS     (NN / NSPLIT)   /* 1024 columns per split */

/* workspace byte offsets (all 16B aligned) */
#define WS_WH    0u          /* float    [NN*OUT_F]        2,097,152 B */
#define WS_WHT   2097152u    /* _Float16 [OUT_F*NN]        1,048,576 B */
#define WS_S1    3145728u    /* float    [NN]                 32,768 B */
#define WS_S2L   3178496u    /* float    [NN]  s2*log2e       32,768 B */
#define WS_S2A   3211264u    /* float    [NN]  s2*log2e*alpha 32,768 B */
#define WS_MAX   3244032u    /* float    [1] (+pad)               16 B */
#define WS_ZP    3244048u    /* float    [NSPLIT*NN]         262,144 B */
#define WS_OACC  3506192u    /* float    [NSPLIT*NN*OUT_F] 16,777,216 B */
/* total: 20,283,408 B (~19.3 MB) */

/* ---- Wh = h @ W (f32). thread = (i,f); h row broadcast, W coalesced ---- */
__global__ void k_wh(const float* __restrict__ h, const float* __restrict__ W,
                     float* __restrict__ Wh) {
  int tid = blockIdx.x * blockDim.x + threadIdx.x;
  int i = tid >> 6, f = tid & 63;
  const float* hr = h + i * IN_F;
  float acc = 0.f;
#pragma unroll 8
  for (int k = 0; k < IN_F; ++k) acc = fmaf(hr[k], W[k * OUT_F + f], acc);
  Wh[tid] = acc;
}

/* ---- s1 = Wh@a1; s2l = log2e*(Wh@a2); s2a = alpha*s2l ---- */
__global__ void k_s(const float* __restrict__ Wh, const float* __restrict__ a,
                    float* __restrict__ s1, float* __restrict__ s2l,
                    float* __restrict__ s2a) {
  int i = blockIdx.x * blockDim.x + threadIdx.x;
  const float* r = Wh + i * OUT_F;
  float x = 0.f, y = 0.f;
#pragma unroll
  for (int f = 0; f < OUT_F; ++f) {
    float v = r[f];
    x = fmaf(v, a[f], x);
    y = fmaf(v, a[OUT_F + f], y);
  }
  s1[i]  = x;
  s2l[i] = y * LOG2E;
  s2a[i] = y * LOG2E * LALPHA;
}

/* ---- global max of s2l (single block) ---- */
__global__ void k_max(const float* __restrict__ s2l, float* __restrict__ mx) {
  __shared__ float red[1024];
  int t = threadIdx.x;
  float m = -3.4e38f;
  for (int k = t; k < NN; k += 1024) m = fmaxf(m, s2l[k]);
  red[t] = m;
  __syncthreads();
  for (int off = 512; off > 0; off >>= 1) {
    if (t < off) red[t] = fmaxf(red[t], red[t + off]);
    __syncthreads();
  }
  if (t == 0) mx[0] = red[0];
}

/* ---- WhT[f][i] = (f16) Wh[i][f]; writes coalesced, reads cache-resident ---- */
__global__ void k_tr(const float* __restrict__ Wh, _Float16* __restrict__ WhT) {
  int tid = blockIdx.x * blockDim.x + threadIdx.x; /* = f*NN + i */
  int f = tid >> 13, i = tid & (NN - 1);
  WhT[tid] = (_Float16)Wh[i * OUT_F + f];
}

/* ---- fused masked-softmax-numerator x Wh GEMM (flash-attention style) ----
 * wave = 16 output rows x 1024 columns. Per 32-col tile:
 *   B (SRC1) = P^T 32x16 f16 : lane n -> column i0+(n&15),
 *       lanes 0-15: j0+0..15 (2 packed f16 / VGPR), lanes 16-31: j0+16..31
 *   A (SRC0) = WhT tile 16(f) x 32(j) f16 : lane L -> M=f=fb*16+(L&15),
 *       K 0-15 (L<16) / 16-31 (L>=16), same order as B
 *   D = outT tile (f x i), f32, 4 feature blocks -> 4x v_wmma_f32_16x16x32_f16
 * Weight (log2 domain, alpha folded):
 *   E = max(P0 + s2l[j], P1 + s2a[j]) <= 0 ;  w = adj ? exp2(E) : 0
 */
__global__ void __launch_bounds__(32)
k_main(const int* __restrict__ adj, const _Float16* __restrict__ WhT,
       const float* __restrict__ s1, const float* __restrict__ s2l,
       const float* __restrict__ s2a, const float* __restrict__ mxp,
       float* __restrict__ oacc, float* __restrict__ zp) {
  const int lane = threadIdx.x;
  const int blk  = blockIdx.x;
  const int s    = blk & (NSPLIT - 1);
  const int i0   = (blk >> 3) * 16;
  const int row  = i0 + (lane & 15);
  const int sub  = lane >> 4;           /* 0: K 0-15, 1: K 16-31 (B);  A: klo 0/16 */
  const int sbase = s * CPS;
  const int myF  = lane & 15;
  const int klo  = sub ? 16 : 0;

  const float p0 = s1[row] * LOG2E;
  const float t0 = p0 + mxp[0];                 /* log2e*(s1_i + max s2) */
  const float C  = fmaxf(t0, LALPHA * t0);      /* log2e * lrelu(...)    */
  const float P0 = p0 - C;
  const float P1 = LALPHA * p0 - C;

  const int*   adjp = adj + (size_t)row * NN + sbase + sub * 16;
  const float* s2lp = s2l + sbase + sub * 16;
  const float* s2ap = s2a + sbase + sub * 16;

  v8f acc[4] = {};
  float z0 = 0.f, z1 = 0.f, z2 = 0.f, z3 = 0.f;

  for (int t = 0; t < CPS / 32; ++t) {
    const int j0 = t * 32;
    const v4i A0 = __builtin_nontemporal_load((const v4i*)(adjp + j0));
    const v4i A1 = __builtin_nontemporal_load((const v4i*)(adjp + j0 + 4));
    const v4i A2 = __builtin_nontemporal_load((const v4i*)(adjp + j0 + 8));
    const v4i A3 = __builtin_nontemporal_load((const v4i*)(adjp + j0 + 12));
    const float4 L0 = *(const float4*)(s2lp + j0);
    const float4 L1 = *(const float4*)(s2lp + j0 + 4);
    const float4 L2 = *(const float4*)(s2lp + j0 + 8);
    const float4 L3 = *(const float4*)(s2lp + j0 + 12);
    const float4 G0 = *(const float4*)(s2ap + j0);
    const float4 G1 = *(const float4*)(s2ap + j0 + 4);
    const float4 G2 = *(const float4*)(s2ap + j0 + 8);
    const float4 G3 = *(const float4*)(s2ap + j0 + 12);
    const int av[16]    = {A0[0],A0[1],A0[2],A0[3], A1[0],A1[1],A1[2],A1[3],
                           A2[0],A2[1],A2[2],A2[3], A3[0],A3[1],A3[2],A3[3]};
    const float svl[16] = {L0.x,L0.y,L0.z,L0.w, L1.x,L1.y,L1.z,L1.w,
                           L2.x,L2.y,L2.z,L2.w, L3.x,L3.y,L3.z,L3.w};
    const float sva[16] = {G0.x,G0.y,G0.z,G0.w, G1.x,G1.y,G1.z,G1.w,
                           G2.x,G2.y,G2.z,G2.w, G3.x,G3.y,G3.z,G3.w};
    float w[16];
#pragma unroll
    for (int k = 0; k < 16; ++k) {
      float E = fmaxf(P0 + svl[k], P1 + sva[k]);          /* <= 0, log2 domain */
      w[k] = (av[k] > 0) ? __builtin_amdgcn_exp2f(E) : 0.f; /* 1x v_exp_f32 */
    }
    v16h b;
#pragma unroll
    for (int k = 0; k < 4; ++k) {   /* 4 independent Z chains (ILP) */
      z0 += w[4 * k + 0];
      z1 += w[4 * k + 1];
      z2 += w[4 * k + 2];
      z3 += w[4 * k + 3];
    }
#pragma unroll
    for (int k = 0; k < 16; ++k) b[k] = (_Float16)w[k];
#pragma unroll
    for (int fb = 0; fb < 4; ++fb) {
      const _Float16* ap = WhT + (size_t)(fb * 16 + myF) * NN + sbase + j0 + klo;
      v8h lo = *(const v8h*)(ap);
      v8h hi = *(const v8h*)(ap + 8);
      v16h Av = __builtin_shufflevector(lo, hi, 0,1,2,3,4,5,6,7,8,9,10,11,12,13,14,15);
      acc[fb] = __builtin_amdgcn_wmma_f32_16x16x32_f16(
          false, Av, false, b, (short)0, acc[fb], false, false);
    }
  }

  /* Z: lanes n and n+16 hold the two halves of row i0+n */
  float zacc = (z0 + z1) + (z2 + z3);
  float zo = zacc + __shfl_xor(zacc, 16, 32);
  if (lane < 16) zp[s * NN + i0 + lane] = zo;

  /* D layout: VGPR v, lane n -> f = fb*16 + (n>=16?8:0) + v, i = i0+(n&15) */
  float* ob = oacc + ((size_t)(s * NN + i0 + (lane & 15))) * OUT_F + sub * 8;
#pragma unroll
  for (int fb = 0; fb < 4; ++fb) {
    v4f lo4 = {acc[fb][0], acc[fb][1], acc[fb][2], acc[fb][3]};
    v4f hi4 = {acc[fb][4], acc[fb][5], acc[fb][6], acc[fb][7]};
    __builtin_nontemporal_store(lo4, (v4f*)(ob + fb * 16));
    __builtin_nontemporal_store(hi4, (v4f*)(ob + fb * 16 + 4));
  }
}

/* ---- reduce splits, normalize, ELU ---- */
__global__ void k_final(const float* __restrict__ oacc, const float* __restrict__ zp,
                        float* __restrict__ out) {
  int tid = blockIdx.x * blockDim.x + threadIdx.x;
  int i = tid >> 6, f = tid & 63;
  float sum = 0.f, z = 0.f;
#pragma unroll
  for (int s = 0; s < NSPLIT; ++s) {
    sum += oacc[((size_t)(s * NN + i)) * OUT_F + f];
    z   += zp[s * NN + i];
  }
  float r = sum / z;
  out[tid] = (r > 0.f) ? r : (__expf(r) - 1.f);
}

extern "C" void kernel_launch(void* const* d_in, const int* in_sizes, int n_in,
                              void* d_out, int out_size, void* d_ws, size_t ws_size,
                              hipStream_t stream) {
  (void)in_sizes; (void)n_in; (void)out_size; (void)ws_size;
  const float* h   = (const float*)d_in[0];
  const int*   adj = (const int*)d_in[1];
  const float* W   = (const float*)d_in[2];
  const float* a   = (const float*)d_in[3];
  float* out = (float*)d_out;

  char* ws = (char*)d_ws;
  float*    Wh  = (float*)(ws + WS_WH);
  _Float16* WhT = (_Float16*)(ws + WS_WHT);
  float*    s1  = (float*)(ws + WS_S1);
  float*    s2l = (float*)(ws + WS_S2L);
  float*    s2a = (float*)(ws + WS_S2A);
  float*    mx  = (float*)(ws + WS_MAX);
  float*    zp  = (float*)(ws + WS_ZP);
  float*    oac = (float*)(ws + WS_OACC);

  k_wh   <<<(NN * OUT_F) / 256, 256, 0, stream>>>(h, W, Wh);
  k_s    <<<NN / 256,           256, 0, stream>>>(Wh, a, s1, s2l, s2a);
  k_max  <<<1,                 1024, 0, stream>>>(s2l, mx);
  k_tr   <<<(NN * OUT_F) / 256, 256, 0, stream>>>(Wh, WhT);
  k_main <<<(NN / 16) * NSPLIT,  32, 0, stream>>>(adj, WhT, s1, s2l, s2a, mx, oac, zp);
  k_final<<<(NN * OUT_F) / 256, 256, 0, stream>>>(oac, zp, out);
}
